// GatedCrossAttentionBlock_86131274154339
// MI455X (gfx1250) — matrix operand order repaired and checked by reading, not compiled, not run on hardware
//
#include <hip/hip_runtime.h>
#include <hip/hip_bf16.h>
#include <math.h>

// Problem constants (from reference)
#define BC     4
#define T1C    2048
#define DIMC   1024
#define DLC    1024
#define DHC    64
#define HC     8
#define INNERC 512          // H*DH
#define JC     512          // TKV*N
#define DFFC   4096

typedef unsigned short bf16_t;
typedef __attribute__((ext_vector_type(16))) __bf16 v16bf;
typedef __attribute__((ext_vector_type(8)))  float  v8f;

union FragAB { v16bf v; uint4 q[2]; };

__device__ __forceinline__ bf16_t f2bf(float f) {
  union { float f; unsigned u; } c; c.f = f;
  unsigned r = c.u + 0x7FFFu + ((c.u >> 16) & 1u);   // round-to-nearest-even
  return (bf16_t)(r >> 16);
}

// ---------------------------------------------------------------------------
// elementwise f32 -> bf16
// ---------------------------------------------------------------------------
__global__ __launch_bounds__(256)
void to_bf16_k(const float* __restrict__ x, bf16_t* __restrict__ y, long n) {
  long i = (long)blockIdx.x * 256 + threadIdx.x;
  if (i < n) y[i] = f2bf(x[i]);
}

// ---------------------------------------------------------------------------
// LayerNorm over DIMC=1024 cols, one block per row, bf16 output
// ---------------------------------------------------------------------------
__global__ __launch_bounds__(256)
void layernorm_k(const float* __restrict__ X, const float* __restrict__ g,
                 const float* __restrict__ b, bf16_t* __restrict__ Y) {
  __shared__ float ssum[256];
  __shared__ float ssq[256];
  const int row = blockIdx.x;
  const int tid = threadIdx.x;
  const float* xr = X + (size_t)row * DIMC;
  float loc[4];
  float s = 0.f, sq = 0.f;
#pragma unroll
  for (int k = 0; k < 4; ++k) {
    float x = xr[tid + k * 256];
    loc[k] = x; s += x; sq += x * x;
  }
  ssum[tid] = s; ssq[tid] = sq;
  __syncthreads();
  for (int off = 128; off > 0; off >>= 1) {
    if (tid < off) { ssum[tid] += ssum[tid + off]; ssq[tid] += ssq[tid + off]; }
    __syncthreads();
  }
  const float mu   = ssum[0] * (1.0f / DIMC);
  const float var  = ssq[0] * (1.0f / DIMC) - mu * mu;
  const float rstd = rsqrtf(var + 1e-5f);
#pragma unroll
  for (int k = 0; k < 4; ++k) {
    int c = tid + k * 256;
    float y = (loc[k] - mu) * rstd * g[c] + b[c];
    Y[(size_t)row * DIMC + c] = f2bf(y);
  }
}

// ---------------------------------------------------------------------------
// bf16 x bf16 -> f32 WMMA GEMM.  C = epilogue(A[MxK] @ B[KxN])
// Block tile 128x128, 8 wave32 waves in 4x2, each wave 32x64 (2x4 WMMA accs).
// K-step 32 with register-prefetch + double-buffered LDS (1 barrier/step).
// MODE 0: f32 out * alpha        MODE 1: bf16 out * alpha
// MODE 2: bf16 out, exact GELU   MODE 3: f32 out = acc*tanh(*gate) + aux
// ---------------------------------------------------------------------------
template <int MODE>
__global__ __launch_bounds__(256)
void gemm_bf16_k(const bf16_t* __restrict__ A, const bf16_t* __restrict__ Bm,
                 int M, int N, int K, float alpha,
                 const float* __restrict__ gate, const float* __restrict__ aux,
                 void* __restrict__ Cout) {
  __shared__ __align__(16) bf16_t As[2][128 * 40];   // [buf][row][k], stride 40
  __shared__ __align__(16) bf16_t Bts[2][128 * 40];  // [buf][n][k] (B transposed)

  const int tid  = threadIdx.x;
  const int lane = tid & 31;
  const int hf   = lane >> 4;      // half-wave id
  const int l16  = lane & 15;
  const int w    = tid >> 5;
  const int wrow = w >> 1;         // 0..3  (rows of 32)
  const int wcol = w & 1;          // 0..1  (cols of 64)
  const int m0   = blockIdx.y * 128;
  const int n0   = blockIdx.x * 128;

  // staging coordinates
  const int ar = tid >> 1;                // 0..127
  const int ac = (tid & 1) * 16;          // 0 or 16
  const int bk = tid >> 4;                // 0..15 (and +16)
  const int bn = (tid & 15) * 8;          // 0..120

  uint4 pa0, pa1, pb0, pb1;
  auto load_tile_regs = [&](int k0) {
    const bf16_t* ga = A + (size_t)(m0 + ar) * K + k0 + ac;
    pa0 = ((const uint4*)ga)[0];
    pa1 = ((const uint4*)ga)[1];
    const bf16_t* gb = Bm + (size_t)(k0 + bk) * N + n0 + bn;
    pb0 = *(const uint4*)gb;
    pb1 = *(const uint4*)(gb + (size_t)16 * N);
  };
  auto store_tile_lds = [&](int buf) {
    *(uint4*)&As[buf][ar * 40 + ac]     = pa0;
    *(uint4*)&As[buf][ar * 40 + ac + 8] = pa1;
    bf16_t t0[8], t1[8];
    *(uint4*)t0 = pb0;
    *(uint4*)t1 = pb1;
#pragma unroll
    for (int e = 0; e < 8; ++e) {
      Bts[buf][(bn + e) * 40 + bk]      = t0[e];
      Bts[buf][(bn + e) * 40 + bk + 16] = t1[e];
    }
  };

  v8f acc[2][4];
#pragma unroll
  for (int a = 0; a < 2; ++a)
#pragma unroll
    for (int b = 0; b < 4; ++b)
      acc[a][b] = (v8f){0.f, 0.f, 0.f, 0.f, 0.f, 0.f, 0.f, 0.f};

  load_tile_regs(0);
  store_tile_lds(0);
  __syncthreads();

  int cur = 0;
  for (int k0 = 0; k0 < K; k0 += 32) {
    const bool has_next = (k0 + 32) < K;
    if (has_next) load_tile_regs(k0 + 32);   // overlap with compute below

    FragAB af[2], bfv[4];
    const int akb = hf ? 8 : 0;     // A halves: K {0..7,16..23}/{8..15,24..31}
#pragma unroll
    for (int rt = 0; rt < 2; ++rt) {
      int row = wrow * 32 + rt * 16 + l16;
      af[rt].q[0] = *(const uint4*)&As[cur][row * 40 + akb];
      af[rt].q[1] = *(const uint4*)&As[cur][row * 40 + akb + 16];
    }
    const int bkb = hf ? 8 : 0;     // B halves: K {0..7,16..23}/{8..15,24..31}
#pragma unroll
    for (int ct = 0; ct < 4; ++ct) {
      int col = wcol * 64 + ct * 16 + l16;
      bfv[ct].q[0] = *(const uint4*)&Bts[cur][col * 40 + bkb];
      bfv[ct].q[1] = *(const uint4*)&Bts[cur][col * 40 + bkb + 16];
    }
#pragma unroll
    for (int rt = 0; rt < 2; ++rt)
#pragma unroll
      for (int ct = 0; ct < 4; ++ct)
        acc[rt][ct] = __builtin_amdgcn_wmma_f32_16x16x32_bf16(
            false, af[rt].v, false, bfv[ct].v, (short)0, acc[rt][ct], false, false);

    if (has_next) {
      store_tile_lds(cur ^ 1);   // writers/readers of this buf fenced last iter
      __syncthreads();
      cur ^= 1;
    }
  }

  float tg = 0.f;
  if constexpr (MODE == 3) tg = tanhf(gate[0]);

#pragma unroll
  for (int rt = 0; rt < 2; ++rt)
#pragma unroll
    for (int ct = 0; ct < 4; ++ct)
#pragma unroll
      for (int i = 0; i < 8; ++i) {
        int r = m0 + wrow * 32 + rt * 16 + (hf ? i + 8 : i);
        int c = n0 + wcol * 64 + ct * 16 + l16;
        size_t idx = (size_t)r * N + c;
        float v = acc[rt][ct][i];
        if constexpr (MODE == 0) {
          ((float*)Cout)[idx] = v * alpha;
        } else if constexpr (MODE == 1) {
          ((bf16_t*)Cout)[idx] = f2bf(v * alpha);
        } else if constexpr (MODE == 2) {
          float gl = 0.5f * v * (1.0f + erff(v * 0.70710678118654752f));
          ((bf16_t*)Cout)[idx] = f2bf(gl);
        } else {
          ((float*)Cout)[idx] = v * tg + aux[idx];
        }
      }
}

// ---------------------------------------------------------------------------
// Attention: per block = (32 q-rows, one head h, one batch b)
//   sim = Qs @ K^T  (WMMA, K frags straight from global kv)
//   masked softmax in registers (masked -> -FLT_MAX, exactly like reference)
//   O = P @ V       (WMMA, V transposed into LDS per 128-j chunk)
// ---------------------------------------------------------------------------
__global__ __launch_bounds__(256)
void attn_k(const bf16_t* __restrict__ Q,          // (B*T1, INNER), pre-scaled
            const bf16_t* __restrict__ KV,         // (B*J, 2*INNER): k | v
            const unsigned char* __restrict__ amask, // (B, T1, J)
            const unsigned char* __restrict__ kmask, // (B, J)
            const float* __restrict__ qmask,         // (B, T1)
            bf16_t* __restrict__ O) {                // (B*T1, INNER)
  __shared__ __align__(16) bf16_t Qs[32 * 72];
  __shared__ __align__(16) bf16_t P[32 * 520];
  __shared__ __align__(16) char scratch[64 * 136 * 2];  // Msk (16KB) then Vt (17KB)
  __shared__ float red[32 * 8];

  unsigned char* Msk = (unsigned char*)scratch;
  bf16_t*        Vt  = (bf16_t*)scratch;

  const int tid  = threadIdx.x;
  const int lane = tid & 31;
  const int hf   = lane >> 4;
  const int l16  = lane & 15;
  const int w    = tid >> 5;
  const int t0   = blockIdx.x * 32;
  const int h    = blockIdx.y;
  const int b    = blockIdx.z;

  // ---- stage Q rows (32 x 64) ----
  {
    int r = tid >> 3;
    int c = (tid & 7) * 8;
    const bf16_t* g = Q + ((size_t)(b * T1C + t0 + r)) * INNERC + h * DHC + c;
    *(uint4*)&Qs[r * 72 + c] = *(const uint4*)g;
  }
  // ---- stage combined mask (attn & kv), 32 x 512 bytes ----
  {
    int r  = tid >> 3;
    int cb = (tid & 7) * 64;
    const uint4* ag = (const uint4*)(amask + ((size_t)b * T1C + t0 + r) * JC + cb);
    const uint4* kg = (const uint4*)(kmask + (size_t)b * JC + cb);
#pragma unroll
    for (int q4 = 0; q4 < 4; ++q4) {
      uint4 a = ag[q4], k = kg[q4];
      uint4 m; m.x = a.x & k.x; m.y = a.y & k.y; m.z = a.z & k.z; m.w = a.w & k.w;
      *(uint4*)&Msk[r * 512 + cb + q4 * 16] = m;
    }
  }
  __syncthreads();

  // ---- phase 1: sim = Q @ K^T; wave w owns columns [w*64, w*64+64) ----
  v8f acc[2][4];
#pragma unroll
  for (int rt = 0; rt < 2; ++rt)
#pragma unroll
    for (int ct = 0; ct < 4; ++ct)
      acc[rt][ct] = (v8f){0.f, 0.f, 0.f, 0.f, 0.f, 0.f, 0.f, 0.f};

  const size_t kvrow0 = (size_t)b * JC;
#pragma unroll
  for (int s = 0; s < 2; ++s) {                     // DH=64 -> 2 K-steps of 32
    FragAB af[2];
    const int akb = s * 32 + (hf ? 8 : 0);
    af[0].q[0] = *(const uint4*)&Qs[l16 * 72 + akb];
    af[0].q[1] = *(const uint4*)&Qs[l16 * 72 + akb + 16];
    af[1].q[0] = *(const uint4*)&Qs[(16 + l16) * 72 + akb];
    af[1].q[1] = *(const uint4*)&Qs[(16 + l16) * 72 + akb + 16];
    const int dhb = s * 32 + (hf ? 8 : 0);
#pragma unroll
    for (int ct = 0; ct < 4; ++ct) {
      int j = w * 64 + ct * 16 + l16;
      const bf16_t* g = KV + (kvrow0 + j) * (2 * INNERC) + h * DHC + dhb;
      FragAB bfv;
      bfv.q[0] = ((const uint4*)g)[0];
      bfv.q[1] = ((const uint4*)g)[2];
#pragma unroll
      for (int rt = 0; rt < 2; ++rt)
        acc[rt][ct] = __builtin_amdgcn_wmma_f32_16x16x32_bf16(
            false, af[rt].v, false, bfv.v, (short)0, acc[rt][ct], false, false);
    }
  }

  // ---- phase 2: mask + softmax (reference uses finite finfo.min) ----
#pragma unroll
  for (int rt = 0; rt < 2; ++rt)
#pragma unroll
    for (int i = 0; i < 8; ++i) {
      int m = rt * 16 + (hf ? i + 8 : i);
#pragma unroll
      for (int ct = 0; ct < 4; ++ct) {
        int j = w * 64 + ct * 16 + l16;
        if (!Msk[m * 512 + j]) acc[rt][ct][i] = -__FLT_MAX__;
      }
    }

  float rmax[2][8];
#pragma unroll
  for (int rt = 0; rt < 2; ++rt)
#pragma unroll
    for (int i = 0; i < 8; ++i) {
      float mx = fmaxf(fmaxf(acc[rt][0][i], acc[rt][1][i]),
                       fmaxf(acc[rt][2][i], acc[rt][3][i]));
#pragma unroll
      for (int off = 1; off < 16; off <<= 1) mx = fmaxf(mx, __shfl_xor(mx, off, 32));
      int m = rt * 16 + (hf ? i + 8 : i);
      if (l16 == 0) red[m * 8 + w] = mx;
    }
  __syncthreads();
#pragma unroll
  for (int rt = 0; rt < 2; ++rt)
#pragma unroll
    for (int i = 0; i < 8; ++i) {
      int m = rt * 16 + (hf ? i + 8 : i);
      float g = red[m * 8];
#pragma unroll
      for (int ww = 1; ww < 8; ++ww) g = fmaxf(g, red[m * 8 + ww]);
      rmax[rt][i] = g;
    }
  __syncthreads();
#pragma unroll
  for (int rt = 0; rt < 2; ++rt)
#pragma unroll
    for (int i = 0; i < 8; ++i) {
      float sm = 0.f;
#pragma unroll
      for (int ct = 0; ct < 4; ++ct) {
        float e = __expf(acc[rt][ct][i] - rmax[rt][i]);
        acc[rt][ct][i] = e;
        sm += e;
      }
#pragma unroll
      for (int off = 1; off < 16; off <<= 1) sm += __shfl_xor(sm, off, 32);
      int m = rt * 16 + (hf ? i + 8 : i);
      if (l16 == 0) red[m * 8 + w] = sm;
    }
  __syncthreads();
#pragma unroll
  for (int rt = 0; rt < 2; ++rt)
#pragma unroll
    for (int i = 0; i < 8; ++i) {
      int m = rt * 16 + (hf ? i + 8 : i);
      float gs = 0.f;
#pragma unroll
      for (int ww = 0; ww < 8; ++ww) gs += red[m * 8 + ww];
      float inv = 1.0f / gs;
#pragma unroll
      for (int ct = 0; ct < 4; ++ct) {
        int j = w * 64 + ct * 16 + l16;
        P[m * 520 + j] = f2bf(acc[rt][ct][i] * inv);
      }
    }

  // ---- phase 3: O = P @ V; wave w owns 16x16 output tile (prt, pct) ----
  const int prt = w >> 2, pct = w & 3;
  v8f o = (v8f){0.f, 0.f, 0.f, 0.f, 0.f, 0.f, 0.f, 0.f};
  for (int jc = 0; jc < JC; jc += 128) {
    __syncthreads();                       // P done / previous Vt consumed
    {
      int jj  = tid >> 1;
      int dhb = (tid & 1) * 32;
      const bf16_t* g = KV + (kvrow0 + jc + jj) * (2 * INNERC) + INNERC + h * DHC + dhb;
      bf16_t tmp[32];
      ((uint4*)tmp)[0] = ((const uint4*)g)[0];
      ((uint4*)tmp)[1] = ((const uint4*)g)[1];
      ((uint4*)tmp)[2] = ((const uint4*)g)[2];
      ((uint4*)tmp)[3] = ((const uint4*)g)[3];
#pragma unroll
      for (int e = 0; e < 32; ++e) Vt[(dhb + e) * 136 + jj] = tmp[e];
    }
    __syncthreads();
#pragma unroll
    for (int s = 0; s < 4; ++s) {
      FragAB a_, b_;
      int akb = jc + s * 32 + (hf ? 8 : 0);
      a_.q[0] = *(const uint4*)&P[(prt * 16 + l16) * 520 + akb];
      a_.q[1] = *(const uint4*)&P[(prt * 16 + l16) * 520 + akb + 16];
      int bkb = s * 32 + (hf ? 8 : 0);
      b_.q[0] = *(const uint4*)&Vt[(pct * 16 + l16) * 136 + bkb];
      b_.q[1] = *(const uint4*)&Vt[(pct * 16 + l16) * 136 + bkb + 16];
      o = __builtin_amdgcn_wmma_f32_16x16x32_bf16(false, a_.v, false, b_.v,
                                                  (short)0, o, false, false);
    }
  }
#pragma unroll
  for (int i = 0; i < 8; ++i) {
    int m  = prt * 16 + (hf ? i + 8 : i);
    int t  = t0 + m;
    int dh = pct * 16 + l16;
    float qm = qmask[(size_t)b * T1C + t];
    O[((size_t)b * T1C + t) * INNERC + h * DHC + dh] = f2bf(o[i] * qm);
  }
}

// ---------------------------------------------------------------------------
// Launcher
// ---------------------------------------------------------------------------
extern "C" void kernel_launch(void* const* d_in, const int* in_sizes, int n_in,
                              void* d_out, int out_size, void* d_ws, size_t ws_size,
                              hipStream_t stream) {
  (void)in_sizes; (void)n_in; (void)out_size; (void)ws_size;
  const float*         qo       = (const float*)d_in[0];
  const float*         kvo      = (const float*)d_in[1];
  const unsigned char* amask    = (const unsigned char*)d_in[2];
  const float*         qmask    = (const float*)d_in[3];
  const unsigned char* kmask    = (const unsigned char*)d_in[4];
  const float*         ln_g     = (const float*)d_in[5];
  const float*         ln_b     = (const float*)d_in[6];
  const float*         Wq       = (const float*)d_in[7];
  const float*         Wkv      = (const float*)d_in[8];
  const float*         Wout     = (const float*)d_in[9];
  const float*         attn_g   = (const float*)d_in[10];
  const float*         ff_ln_g  = (const float*)d_in[11];
  const float*         ff_ln_b  = (const float*)d_in[12];
  const float*         W1       = (const float*)d_in[13];
  const float*         W2       = (const float*)d_in[14];
  const float*         ff_gate  = (const float*)d_in[15];
  float*               out      = (float*)d_out;

  size_t off = 0;
  auto alloc = [&](size_t nbytes) -> void* {
    void* r = (char*)d_ws + off;
    off += (nbytes + 255) & ~(size_t)255;
    return r;
  };
  const size_t ROWS = (size_t)BC * T1C;    // 8192
  const size_t KVR  = (size_t)BC * JC;     // 2048

  bf16_t* Wq_b   = (bf16_t*)alloc((size_t)DIMC * INNERC * 2);
  bf16_t* Wkv_b  = (bf16_t*)alloc((size_t)DLC * 2 * INNERC * 2);
  bf16_t* Wout_b = (bf16_t*)alloc((size_t)INNERC * DIMC * 2);
  bf16_t* W1_b   = (bf16_t*)alloc((size_t)DIMC * DFFC * 2);
  bf16_t* W2_b   = (bf16_t*)alloc((size_t)DFFC * DIMC * 2);
  bf16_t* kvo_b  = (bf16_t*)alloc(KVR * DLC * 2);
  bf16_t* xln_b  = (bf16_t*)alloc(ROWS * DIMC * 2);
  bf16_t* q_b    = (bf16_t*)alloc(ROWS * INNERC * 2);
  bf16_t* kv_b   = (bf16_t*)alloc(KVR * 2 * INNERC * 2);
  bf16_t* O_b    = (bf16_t*)alloc(ROWS * INNERC * 2);
  float*  x_f    = (float*)alloc(ROWS * DIMC * 4);
  bf16_t* xln2_b = (bf16_t*)alloc(ROWS * DIMC * 2);
  bf16_t* h_b    = (bf16_t*)alloc(ROWS * DFFC * 2);

  auto cvt = [&](const float* src, bf16_t* dst, long n) {
    to_bf16_k<<<(unsigned)((n + 255) / 256), 256, 0, stream>>>(src, dst, n);
  };
  cvt(Wq,   Wq_b,   (long)DIMC * INNERC);
  cvt(Wkv,  Wkv_b,  (long)DLC * 2 * INNERC);
  cvt(Wout, Wout_b, (long)INNERC * DIMC);
  cvt(W1,   W1_b,   (long)DIMC * DFFC);
  cvt(W2,   W2_b,   (long)DFFC * DIMC);
  cvt(kvo,  kvo_b,  (long)KVR * DLC);

  // ln(qo) -> bf16
  layernorm_k<<<(unsigned)ROWS, 256, 0, stream>>>(qo, ln_g, ln_b, xln_b);

  // q = ln(qo) @ Wq, scaled by DH^-0.5, bf16 out
  gemm_bf16_k<1><<<dim3(INNERC / 128, ROWS / 128), 256, 0, stream>>>(
      xln_b, Wq_b, (int)ROWS, INNERC, DIMC, 0.125f, nullptr, nullptr, q_b);

  // kv = kv_flat @ Wkv, bf16 out (k | v)
  gemm_bf16_k<1><<<dim3((2 * INNERC) / 128, KVR / 128), 256, 0, stream>>>(
      kvo_b, Wkv_b, (int)KVR, 2 * INNERC, DLC, 1.0f, nullptr, nullptr, kv_b);

  // masked softmax attention
  attn_k<<<dim3(T1C / 32, HC, BC), 256, 0, stream>>>(q_b, kv_b, amask, kmask,
                                                     qmask, O_b);

  // x = (O @ Wout) * tanh(attn_gate) + qo
  gemm_bf16_k<3><<<dim3(DIMC / 128, ROWS / 128), 256, 0, stream>>>(
      O_b, Wout_b, (int)ROWS, DIMC, INNERC, 1.0f, attn_g, qo, x_f);

  // ln(x) -> bf16
  layernorm_k<<<(unsigned)ROWS, 256, 0, stream>>>(x_f, ff_ln_g, ff_ln_b, xln2_b);

  // h = gelu(ln(x) @ W1), bf16 out
  gemm_bf16_k<2><<<dim3(DFFC / 128, ROWS / 128), 256, 0, stream>>>(
      xln2_b, W1_b, (int)ROWS, DFFC, DIMC, 1.0f, nullptr, nullptr, h_b);

  // out = (h @ W2) * tanh(ff_gate) + x
  gemm_bf16_k<3><<<dim3(DIMC / 128, ROWS / 128), 256, 0, stream>>>(
      h_b, W2_b, (int)ROWS, DIMC, DFFC, 1.0f, ff_gate, x_f, out);
}
